// MessageContract_1571958030834
// MI455X (gfx1250) — matrix operand order repaired and checked by reading, not compiled, not run on hardware
//
#include <hip/hip_runtime.h>
#include <hip/hip_bf16.h>

typedef __attribute__((ext_vector_type(16))) _Float16 v16h;
typedef __attribute__((ext_vector_type(8)))  _Float16 v8h;
typedef __attribute__((ext_vector_type(8)))  float    v8f;
typedef __attribute__((ext_vector_type(4)))  float    v4f;

#define CHAN 128
#define RANK 64
#define DE   64

// Compiler fence on a uniform SGPR value: re-defines it each execution so
// LICM/CSE cannot hoist or cache loads whose address depends on it, while
// keeping the base pointer transparent to address-space inference
// (=> global_load with SGPR base, not flat_load with VGPR address pairs).
#define FENCE_S(x) asm volatile("" : "+s"(x))

// ---------------- WMMA helpers ----------------

__device__ __forceinline__ v8f wmma16(v16h a, v16h b, v8f c) {
  // D = A(16x32 f16) * B(32x16 f16) + C(16x16 f32)
  return __builtin_amdgcn_wmma_f32_16x16x32_f16(
      /*neg_a=*/false, a, /*neg_b=*/false, b,
      /*c_mod=*/(short)0, c, /*reuse_a=*/false, /*reuse_b=*/false);
}

// A-fragment (16x32) from global f32 row-major; base points at (row 0, col k0),
// row stride `ld` floats. lane -> m = lane&15 (row), hi = lane>>4.
// Layout: elems 0..7 -> K = 8*hi + e ; elems 8..15 -> K = 16 + 8*hi + (e-8).
__device__ __forceinline__ v16h ldA_g(const float* __restrict__ base, int ld, int m, int hi) {
  const float* p = base + m * ld + 8 * hi;
  v4f x0 = *(const v4f*)(p);
  v4f x1 = *(const v4f*)(p + 4);
  v4f x2 = *(const v4f*)(p + 16);
  v4f x3 = *(const v4f*)(p + 20);
  v16h a;
#pragma unroll
  for (int i = 0; i < 4; i++) {
    a[i]      = (_Float16)x0[i];
    a[4 + i]  = (_Float16)x1[i];
    a[8 + i]  = (_Float16)x2[i];
    a[12 + i] = (_Float16)x3[i];
  }
  return a;
}

// A-fragment (16x32) from LDS f16 row-major, row stride `ld` halfs.
__device__ __forceinline__ v16h ldA_s(const _Float16* base, int ld, int m, int hi) {
  const _Float16* p = base + m * ld + 8 * hi;
  v8h lo = *(const v8h*)(p);
  v8h hv = *(const v8h*)(p + 16);
  v16h a;
#pragma unroll
  for (int i = 0; i < 8; i++) { a[i] = lo[i]; a[8 + i] = hv[i]; }
  return a;
}

// B-fragment (32x16) from f16 weight stored (N,K) row-major; base at (n0, k0),
// row stride `ld` halfs. lane -> n = lane&15 (col), hi = lane>>4.
// Layout: elems 0..7 -> K = 8*hi + e ; elems 8..15 -> K = 16 + 8*hi + (e-8).
__device__ __forceinline__ v16h ldB(const _Float16* __restrict__ base, int ld, int n, int hi) {
  const _Float16* p = base + n * ld + 8 * hi;
  v8h lo = *(const v8h*)(p);
  v8h hv = *(const v8h*)(p + 16);
  v16h b;
#pragma unroll
  for (int i = 0; i < 8; i++) { b[i] = lo[i]; b[8 + i] = hv[i]; }
  return b;
}

// ---------------- weight f32 -> f16 conversion ----------------

__global__ void cvt_f16_kernel(const float* __restrict__ src, _Float16* __restrict__ dst, int n) {
  int i = blockIdx.x * 256 + threadIdx.x;
  if (i < n) dst[i] = (_Float16)src[i];
}

// ---------------- fused message-contract kernel ----------------

__global__ __launch_bounds__(128) void mc_main(
    const float* __restrict__ z0, const float* __restrict__ z1,
    const float* __restrict__ dir, const float* __restrict__ de,
    const float* __restrict__ b0a, const float* __restrict__ bd1,
    const float* __restrict__ bd2, const float* __restrict__ bd3,
    const float* __restrict__ bo0,
    const _Float16* __restrict__ w0a, const _Float16* __restrict__ w1a,
    const _Float16* __restrict__ wd1, const _Float16* __restrict__ wd2,
    const _Float16* __restrict__ wd3, const _Float16* __restrict__ wo0,
    const _Float16* __restrict__ wo1,
    float* __restrict__ out0, float* __restrict__ out1, int nTiles)
{
  __shared__ _Float16 xs[4][16 * 256];   // per-wave transpose buffer (8 KB)
  __shared__ float    shs[4][16 * 12];   // per-wave SH table

  const int lane = threadIdx.x & 31;
  const int wv   = threadIdx.x >> 5;
  const int m    = lane & 15;
  const int hi   = lane >> 4;
  _Float16* xp  = &xs[wv][0];
  float*    shb = &shs[wv][0];

  const int wid = blockIdx.x * 4 + wv;
  const int nw  = gridDim.x * 4;

  const float LRS = 0.1f;
  const float S3   = 1.7320508075688772f;   // sqrt(3)
  const float S15  = 3.872983346207417f;    // sqrt(15)
  const float S5H  = 1.118033988749895f;    // sqrt(5)/2
  const float S15H = 1.9364916731037085f;   // sqrt(15)/2
  const float I15  = 0.2581988897471611f;   // 1/sqrt(15)
  const float I35  = 0.14907119849998599f;  // 1/(3 sqrt(5))
  const float IR3  = 0.5773502691896258f;   // 1/sqrt(3)
  const float IR2  = 0.7071067811865476f;   // 1/sqrt(2)

  for (int tile = wid; tile < nTiles; tile += nw) {
    const int e0 = tile * 16;

    // Loop-"variant" zero offset: blocks LICM from hoisting (and then
    // spilling) the ~100 weight B-fragments, without obscuring the global
    // address space of the weight pointers.
    int wofs = 0;
    FENCE_S(wofs);

    // Prefetch next tile's streaming inputs into the cache hierarchy.
    if (tile + nw < nTiles) {
      const int en = (tile + nw) * 16;
      __builtin_prefetch(z0 + (size_t)en * CHAN + lane * 16, 0, 3);
      __builtin_prefetch(z1 + (size_t)en * 3 * CHAN + lane * 48, 0, 3);
      __builtin_prefetch(de + (size_t)en * DE + lane * 8, 0, 3);
    }

    // ---- spherical harmonics of dir -> LDS table (row e: 9 values) ----
    if (lane < 16) {
      const float x = dir[3 * (e0 + lane) + 0];
      const float y = dir[3 * (e0 + lane) + 1];
      const float z = dir[3 * (e0 + lane) + 2];
      float* s = shb + lane * 12;
      s[0] = 1.0f;
      s[1] = S3 * x;  s[2] = S3 * y;  s[3] = S3 * z;
      s[4] = S15 * x * y;
      s[5] = S15 * y * z;
      s[6] = S5H * (2.f * z * z - x * x - y * y);
      s[7] = S15 * x * z;
      s[8] = S15H * (x * x - y * y);
    }
    __builtin_amdgcn_wave_barrier();

    // ---- distance MLP: de -> h1 -> h2 -> d (D-layout fragments dj) ----
    v8f dj[4];
    {
      v16h af0 = ldA_g(de + (size_t)e0 * DE + 0, DE, m, hi);
      v16h af1 = ldA_g(de + (size_t)e0 * DE + 32, DE, m, hi);
#pragma unroll
      for (int j = 0; j < 4; j++) {
        v8f acc = {};
        acc = wmma16(af0, ldB(wd1 + wofs + (16 * j) * DE + 0, DE, m, hi), acc);
        acc = wmma16(af1, ldB(wd1 + wofs + (16 * j) * DE + 32, DE, m, hi), acc);
        const float bb = bd1[16 * j + m];
#pragma unroll
        for (int i = 0; i < 8; i++) {
          float t = acc[i] + bb;
          t = fmaxf(t, LRS * t);
          xp[(i + 8 * hi) * RANK + 16 * j + m] = (_Float16)t;
        }
      }
      __builtin_amdgcn_wave_barrier();

      v16h bf0 = ldA_s(xp + 0, RANK, m, hi);
      v16h bf1 = ldA_s(xp + 32, RANK, m, hi);
      __builtin_amdgcn_wave_barrier();
      v8f t2[4];
#pragma unroll
      for (int j = 0; j < 4; j++) {
        v8f acc = {};
        acc = wmma16(bf0, ldB(wd2 + wofs + (16 * j) * RANK + 0, RANK, m, hi), acc);
        acc = wmma16(bf1, ldB(wd2 + wofs + (16 * j) * RANK + 32, RANK, m, hi), acc);
        const float bb = bd2[16 * j + m];
#pragma unroll
        for (int i = 0; i < 8; i++) {
          float t = acc[i] + bb;
          t2[j][i] = fmaxf(t, LRS * t);
        }
      }
#pragma unroll
      for (int j = 0; j < 4; j++)
#pragma unroll
        for (int i = 0; i < 8; i++)
          xp[(i + 8 * hi) * RANK + 16 * j + m] = (_Float16)t2[j][i];
      __builtin_amdgcn_wave_barrier();

      v16h cf0 = ldA_s(xp + 0, RANK, m, hi);
      v16h cf1 = ldA_s(xp + 32, RANK, m, hi);
#pragma unroll
      for (int j = 0; j < 4; j++) {
        v8f acc = {};
        acc = wmma16(cf0, ldB(wd3 + wofs + (16 * j) * RANK + 0, RANK, m, hi), acc);
        acc = wmma16(cf1, ldB(wd3 + wofs + (16 * j) * RANK + 32, RANK, m, hi), acc);
        const float bb = bd3[16 * j + m];
#pragma unroll
        for (int i = 0; i < 8; i++) acc[i] += bb;
        dj[j] = acc;
      }
    }

    // ---- a0 = z0 @ W0a^T + b0a ----
    v8f a0j[4];
    {
      v16h za[4];
#pragma unroll
      for (int kk = 0; kk < 4; kk++)
        za[kk] = ldA_g(z0 + (size_t)e0 * CHAN + 32 * kk, CHAN, m, hi);
#pragma unroll
      for (int j = 0; j < 4; j++) {
        v8f acc = {};
#pragma unroll
        for (int kk = 0; kk < 4; kk++)
          acc = wmma16(za[kk], ldB(w0a + wofs + (16 * j) * CHAN + 32 * kk, CHAN, m, hi), acc);
        const float bb = b0a[16 * j + m];
#pragma unroll
        for (int i = 0; i < 8; i++) acc[i] += bb;
        a0j[j] = acc;
      }
    }

    // ---- a1[v] = z1[:,v,:] @ W1a^T ----
    v8f a1j[3][4];
#pragma unroll
    for (int v = 0; v < 3; v++) {
      int vofs = 0;
      FENCE_S(vofs);   // keep the 3 reuses of w1a fragments from being cached
      v16h za[4];
#pragma unroll
      for (int kk = 0; kk < 4; kk++)
        za[kk] = ldA_g(z1 + ((size_t)e0 * 3 + v) * CHAN + 32 * kk, 3 * CHAN, m, hi);
#pragma unroll
      for (int j = 0; j < 4; j++) {
        v8f acc = {};
#pragma unroll
        for (int kk = 0; kk < 4; kk++)
          acc = wmma16(za[kk], ldB(w1a + vofs + (16 * j) * CHAN + 32 * kk, CHAN, m, hi), acc);
        a1j[v][j] = acc;
      }
    }

    // ---- build o0 (16x128 f16) in LDS ----
    __builtin_amdgcn_wave_barrier();
#pragma unroll
    for (int i = 0; i < 8; i++) {
      const int e_i = i + 8 * hi;
      const float* s = shb + e_i * 12;
      const float s0 = s[0], s1 = s[1], s2 = s[2], s3 = s[3];
#pragma unroll
      for (int j = 0; j < 4; j++) {
        const float D = dj[j][i];
        xp[e_i * CHAN + 16 * j + m] = (_Float16)(a0j[j][i] * s0 * D);
        const float sum3 = a1j[0][j][i] * s1 + a1j[1][j][i] * s2 + a1j[2][j][i] * s3;
        xp[e_i * CHAN + 64 + 16 * j + m] = (_Float16)(sum3 * D * IR3);
      }
    }
    __builtin_amdgcn_wave_barrier();

    // ---- out0 = o0 @ Wo0^T + bo0 ----
    {
      v16h af[4];
#pragma unroll
      for (int kk = 0; kk < 4; kk++) af[kk] = ldA_s(xp + 32 * kk, CHAN, m, hi);
#pragma unroll
      for (int j = 0; j < 8; j++) {
        v8f acc = {};
#pragma unroll
        for (int kk = 0; kk < 4; kk++)
          acc = wmma16(af[kk], ldB(wo0 + wofs + (16 * j) * CHAN + 32 * kk, CHAN, m, hi), acc);
        const float bb = bo0[16 * j + m];
#pragma unroll
        for (int i = 0; i < 8; i++)
          out0[(size_t)(e0 + i + 8 * hi) * CHAN + 16 * j + m] = acc[i] + bb;
      }
    }

    // ---- o1[v] (16x256 f16) in LDS, then out1[:,v,:] = o1[v] @ Wo1^T ----
#pragma unroll
    for (int v = 0; v < 3; v++) {
      const int v1 = (v + 1) % 3, v2 = (v + 2) % 3;
      // wo1 fragments are reused for v=0,1,2 (64 fragments = 512 VGPRs if
      // cached); a fresh fenced offset per iteration prevents that.
      int vofs = 0;
      FENCE_S(vofs);
      __builtin_amdgcn_wave_barrier();
#pragma unroll
      for (int i = 0; i < 8; i++) {
        const int e_i = i + 8 * hi;
        const float* s = shb + e_i * 12;
        const float s0 = s[0];
        const float sv  = s[1 + v];
        const float sv1 = s[1 + v1];
        const float sv2 = s[1 + v2];
        const float t0 = s[4], t1 = s[5], t2 = s[6], t3 = s[7], t4 = s[8];
        // _l2mat(c2) row v, with the common factor d[e,r] pulled out.
        const float xy = I15 * t0, yz = I15 * t1, xz = I15 * t3;
        const float zz = 2.f * I35 * t2;
        const float xx =  I15 * t4 - I35 * t2;
        const float yy = -I15 * t4 - I35 * t2;
        const float M0 = (v == 0) ? xx : (v == 1) ? xy : xz;
        const float M1 = (v == 0) ? xy : (v == 1) ? yy : yz;
        const float M2 = (v == 0) ? xz : (v == 1) ? yz : zz;
#pragma unroll
        for (int j = 0; j < 4; j++) {
          const float D   = dj[j][i];
          const float A1x = a1j[0][j][i], A1y = a1j[1][j][i], A1z = a1j[2][j][i];
          const float Av  = (v  == 0) ? A1x : (v  == 1) ? A1y : A1z;
          const float Aa  = (v1 == 0) ? A1x : (v1 == 1) ? A1y : A1z;
          const float Ab  = (v2 == 0) ? A1x : (v2 == 1) ? A1y : A1z;
          _Float16* row = xp + e_i * 256 + 16 * j + m;
          row[0]   = (_Float16)(a0j[j][i] * sv * D);                    // a0 * c1_v
          row[64]  = (_Float16)(Av * s0 * D);                           // a1_v * c0
          row[128] = (_Float16)((Aa * sv2 - Ab * sv1) * D * IR2);       // cross/sqrt2
          row[192] = (_Float16)((M0 * A1x + M1 * A1y + M2 * A1z) * D);  // l2mat . a1
        }
      }
      __builtin_amdgcn_wave_barrier();
#pragma unroll
      for (int j = 0; j < 8; j++) {
        v8f acc = {};
#pragma unroll
        for (int kk = 0; kk < 8; kk++) {
          v16h a = ldA_s(xp + 32 * kk, 256, m, hi);
          acc = wmma16(a, ldB(wo1 + vofs + (16 * j) * 256 + 32 * kk, 256, m, hi), acc);
        }
#pragma unroll
        for (int i = 0; i < 8; i++)
          out1[((size_t)(e0 + i + 8 * hi) * 3 + v) * CHAN + 16 * j + m] = acc[i];
      }
    }
  }
}

// ---------------- host launch ----------------

extern "C" void kernel_launch(void* const* d_in, const int* in_sizes, int n_in,
                              void* d_out, int out_size, void* d_ws, size_t ws_size,
                              hipStream_t stream) {
  (void)n_in; (void)out_size; (void)ws_size;
  const float* z0  = (const float*)d_in[0];
  const float* z1  = (const float*)d_in[1];
  const float* dir = (const float*)d_in[2];
  const float* de  = (const float*)d_in[3];
  const float* W0a = (const float*)d_in[4];
  const float* b0a = (const float*)d_in[5];
  const float* W1a = (const float*)d_in[6];
  const float* Wd1 = (const float*)d_in[7];
  const float* bd1 = (const float*)d_in[8];
  const float* Wd2 = (const float*)d_in[9];
  const float* bd2 = (const float*)d_in[10];
  const float* Wd3 = (const float*)d_in[11];
  const float* bd3 = (const float*)d_in[12];
  const float* Wo0 = (const float*)d_in[13];
  const float* bo0 = (const float*)d_in[14];
  const float* Wo1 = (const float*)d_in[15];

  const int E = in_sizes[0] / CHAN;      // 200000
  const int nTiles = E / 16;             // 12500

  _Float16* wb = (_Float16*)d_ws;
  _Float16* w0a = wb;
  _Float16* w1a = wb + 8192;
  _Float16* wd1 = wb + 16384;
  _Float16* wd2 = wb + 20480;
  _Float16* wd3 = wb + 24576;
  _Float16* wo0 = wb + 28672;
  _Float16* wo1 = wb + 45056;

  cvt_f16_kernel<<<(8192 + 255) / 256, 256, 0, stream>>>(W0a, w0a, 8192);
  cvt_f16_kernel<<<(8192 + 255) / 256, 256, 0, stream>>>(W1a, w1a, 8192);
  cvt_f16_kernel<<<(4096 + 255) / 256, 256, 0, stream>>>(Wd1, wd1, 4096);
  cvt_f16_kernel<<<(4096 + 255) / 256, 256, 0, stream>>>(Wd2, wd2, 4096);
  cvt_f16_kernel<<<(4096 + 255) / 256, 256, 0, stream>>>(Wd3, wd3, 4096);
  cvt_f16_kernel<<<(16384 + 255) / 256, 256, 0, stream>>>(Wo0, wo0, 16384);
  cvt_f16_kernel<<<(32768 + 255) / 256, 256, 0, stream>>>(Wo1, wo1, 32768);

  float* out0 = (float*)d_out;
  float* out1 = out0 + (size_t)E * CHAN;

  mc_main<<<1024, 128, 0, stream>>>(z0, z1, dir, de, b0a, bd1, bd2, bd3, bo0,
                                    w0a, w1a, wd1, wd2, wd3, wo0, wo1,
                                    out0, out1, nTiles);
}
